// GQSWAttention_75900662055278
// MI455X (gfx1250) — matrix operand order repaired and checked by reading, not compiled, not run on hardware
//
#include <hip/hip_runtime.h>
#include <stdint.h>

// ---------------- problem constants ----------------
#define B_     2
#define L_     4096
#define IDIM_  2048
#define NH_    16
#define NG_    4
#define GS_    4
#define HD_    128
#define WIN_   1024
#define SCALE_ 0.0625f      // 256^-0.5
#define EPS_   1e-6f
#define BL_    (B_ * L_)    // 8192 rows

// ---------------- WMMA types ----------------
typedef __attribute__((ext_vector_type(16))) __bf16       v16bf;
typedef __attribute__((ext_vector_type(8)))  float        v8f;
typedef __attribute__((ext_vector_type(4)))  unsigned int u32x4;
typedef __attribute__((ext_vector_type(8)))  unsigned int u32x8;

union FragU { u32x4 u[2]; v16bf v; };

__device__ __forceinline__ unsigned short f2bf(float f) {
    union { float f; unsigned u; } x; x.f = f;
    unsigned r = x.u + 0x7FFFu + ((x.u >> 16) & 1u);
    return (unsigned short)(r >> 16);
}

// A-fragment: 16(M) x 32(K) bf16, source row-major, row stride ld (elements).
__device__ __forceinline__ v16bf load_afrag(const unsigned short* p, int ld) {
    const int lane = threadIdx.x & 31;
    const int m = lane & 15, h = lane >> 4;
    FragU f;
    f.u[0] = *(const u32x4*)(p + m * ld + h * 8);
    f.u[1] = *(const u32x4*)(p + m * ld + 16 + h * 8);
    return f.v;
}

// B-fragment: 32(K) x 16(N) bf16; row n of p holds K-contiguous column n.
__device__ __forceinline__ v16bf load_bfrag(const unsigned short* p, int ld) {
    const int lane = threadIdx.x & 31;
    const int n = lane & 15, h = lane >> 4;
    FragU f;
    f.u[0] = *(const u32x4*)(p + n * ld + h * 8);
    f.u[1] = *(const u32x4*)(p + n * ld + 16 + h * 8);
    return f.v;
}

__device__ __forceinline__ v8f wmma_bf16(v16bf a, v16bf b, v8f c) {
    return __builtin_amdgcn_wmma_f32_16x16x32_bf16(false, a, false, b,
                                                   (short)0, c, false, false);
}

// ---------------- Tensor Data Mover: 2D tile load global -> LDS ----------------
// D# group1 flags: data_size=2B, pad_enable, pad_interval/pad_amount per tile width.
// width 32 bf16 (64B row = 16 DW) + 4 DW pad  -> LDS stride 40 elements
#define TDM_FLAGS_W32  ((1u << 16) | (1u << 20) | (3u << 22) | (3u << 25))
// width 128 bf16 (256B row = 64 DW) + 4 DW pad -> LDS stride 136 elements
#define TDM_FLAGS_W128 ((1u << 16) | (1u << 20) | (5u << 22) | (3u << 25))

__device__ __forceinline__ void tdm_load_2d(unsigned lds_addr, const unsigned short* gaddr,
                                            unsigned tile_w, unsigned rows,
                                            unsigned stride, unsigned flags) {
    unsigned long long ga = (unsigned long long)(uintptr_t)gaddr;
    u32x4 g0;
    g0[0] = 1u;                                   // count=1, user descriptor
    g0[1] = lds_addr;                             // LDS byte address
    g0[2] = (unsigned)ga;                         // global_addr[31:0]
    g0[3] = ((unsigned)(ga >> 32) & 0x01FFFFFFu) | (2u << 30);  // addr[56:32] | type=2
    u32x8 g1;
    g1[0] = flags;                                // mask=0|data_size|pad cfg
    g1[1] = (stride & 0xFFFFu) << 16;             // tensor_dim0 = stride (lo16)
    g1[2] = (stride >> 16) | ((rows & 0xFFFFu) << 16);   // dim0 hi | tensor_dim1 lo
    g1[3] = (rows >> 16) | (tile_w << 16);        // tensor_dim1 hi | tile_dim0
    g1[4] = rows;                                 // tile_dim1 | tile_dim2=0
    g1[5] = stride;                               // tensor_dim0_stride[31:0]
    g1[6] = 0u;                                   // stride hi | dim1_stride lo
    g1[7] = 0u;
    asm volatile("tensor_load_to_lds %0, %1" :: "s"(g0), "s"(g1) : "memory");
}

// ---------------- conversion kernels ----------------
__global__ __launch_bounds__(256) void k_f2bf(const float* __restrict__ in,
                                              unsigned short* __restrict__ out, int n) {
    int i = blockIdx.x * 256 + threadIdx.x;
    if (i < n) out[i] = f2bf(in[i]);
}

// tiled transpose: in [R][C] f32 -> out [C][R] bf16 (both sides coalesced)
__global__ __launch_bounds__(256) void k_transpose_f2bf(const float* __restrict__ in,
                                                        unsigned short* __restrict__ out,
                                                        int R, int C) {
    __shared__ float tile[64][65];
    const int t = threadIdx.x;
    const int tc = t & 63, tr4 = t >> 6;
    const int r0 = blockIdx.y * 64, c0 = blockIdx.x * 64;
#pragma unroll
    for (int i = 0; i < 64; i += 4)
        tile[tr4 + i][tc] = in[(size_t)(r0 + tr4 + i) * C + c0 + tc];
    __syncthreads();
#pragma unroll
    for (int i = 0; i < 64; i += 4)
        out[(size_t)(c0 + tr4 + i) * R + r0 + tc] = f2bf(tile[tc][tr4 + i]);
}

// ---------------- GEMM: C[M][N] f32 = A[M][K] bf16 * Bt[N][K] bf16 ----------------
#define BM 128
#define BN 128
#define BK 32
#define LDT 40   // padded K-stride in LDS (80B)

__global__ __launch_bounds__(256) void k_gemm_bf16_nt(const unsigned short* __restrict__ A,
                                                      const unsigned short* __restrict__ Bt,
                                                      float* __restrict__ C,
                                                      int M, int N, int K) {
    __shared__ __align__(16) unsigned short As[2][BM * LDT];
    __shared__ __align__(16) unsigned short Bs[2][BN * LDT];

    const int t    = threadIdx.x;
    const int lane = t & 31;
    const int hh = lane >> 4, nn = lane & 15;
    const int wid = t >> 5;
    const int w0  = __builtin_amdgcn_readfirstlane(t) >> 5;   // uniform wave id
    const int m0 = blockIdx.y * BM;
    const int n0 = blockIdx.x * BN;
    const int wm = (wid & 3) * 32;   // 4 waves in M
    const int wn = (wid >> 2) * 64;  // 2 waves in N

    const unsigned as_lds = (unsigned)(uintptr_t)(void*)&As[0][0];
    const unsigned bs_lds = (unsigned)(uintptr_t)(void*)&Bs[0][0];
    const unsigned short* agp = A + (size_t)m0 * K;
    const unsigned short* bgp = Bt + (size_t)n0 * K;

    v8f acc[2][4];
#pragma unroll
    for (int i = 0; i < 2; ++i)
#pragma unroll
        for (int j = 0; j < 4; ++j)
#pragma unroll
            for (int e = 0; e < 8; ++e) acc[i][j][e] = 0.f;

    if (w0 == 0) {  // prologue: DMA tile k0=0 into buffer 0
        tdm_load_2d(as_lds, agp, BK, BM, K, TDM_FLAGS_W32);
        tdm_load_2d(bs_lds, bgp, BK, BN, K, TDM_FLAGS_W32);
    }

    int cur = 0;
    for (int k0 = 0; k0 < K; k0 += BK) {
        if (w0 == 0) {
            if (k0 + BK < K) {   // DMA next tile into other buffer, then wait current
                tdm_load_2d(as_lds + (cur ^ 1) * (BM * LDT * 2), agp + k0 + BK, BK, BM, K, TDM_FLAGS_W32);
                tdm_load_2d(bs_lds + (cur ^ 1) * (BN * LDT * 2), bgp + k0 + BK, BK, BN, K, TDM_FLAGS_W32);
                __builtin_amdgcn_s_wait_tensorcnt(2);
            } else {
                __builtin_amdgcn_s_wait_tensorcnt(0);
            }
        }
        __syncthreads();   // tile `cur` ready for everyone

        const unsigned short* as = As[cur];
        const unsigned short* bs = Bs[cur];
        v16bf a[2], b[4];
#pragma unroll
        for (int i = 0; i < 2; ++i) a[i] = load_afrag(as + (wm + i * 16) * LDT, LDT);
#pragma unroll
        for (int j = 0; j < 4; ++j) b[j] = load_bfrag(bs + (wn + j * 16) * LDT, LDT);
#pragma unroll
        for (int i = 0; i < 2; ++i)
#pragma unroll
            for (int j = 0; j < 4; ++j) acc[i][j] = wmma_bf16(a[i], b[j], acc[i][j]);

        __syncthreads();   // all reads of `cur` done before it is DMA-overwritten
        cur ^= 1;
    }

    // 32-bit offset epilogue (SADDR + voffset addressing)
    const unsigned cb = (unsigned)((m0 + wm + 8 * hh) * N + n0 + wn + nn);
#pragma unroll
    for (int i = 0; i < 2; ++i)
#pragma unroll
        for (int j = 0; j < 4; ++j)
#pragma unroll
            for (int r = 0; r < 8; ++r)
                C[cb + (unsigned)((i * 16 + r) * N + j * 16)] = acc[i][j][r];
}

// ---------------- RMSNorm + RoPE ----------------
__device__ __forceinline__ void rms_rope_wave(const float* __restrict__ x,
                                              const float* __restrict__ c,
                                              const float* __restrict__ s,
                                              const float* __restrict__ w,
                                              unsigned short* __restrict__ out) {
    const int lane = threadIdx.x & 31;
    float x0 = x[lane], x1 = x[lane + 32], x2 = x[lane + 64], x3 = x[lane + 96];
    float ss = x0 * x0 + x1 * x1 + x2 * x2 + x3 * x3;
#pragma unroll
    for (int off = 16; off >= 1; off >>= 1) ss += __shfl_xor(ss, off);
    float inv = rsqrtf(ss * (1.f / 128.f) + EPS_);
    x0 *= inv * w[lane];      x1 *= inv * w[lane + 32];
    x2 *= inv * w[lane + 64]; x3 *= inv * w[lane + 96];
    out[lane]      = f2bf(x0 * c[lane]      - x2 * s[lane]);
    out[lane + 32] = f2bf(x1 * c[lane + 32] - x3 * s[lane + 32]);
    out[lane + 64] = f2bf(x2 * c[lane + 64] + x0 * s[lane + 64]);
    out[lane + 96] = f2bf(x3 * c[lane + 96] + x1 * s[lane + 96]);
}

__global__ __launch_bounds__(256) void k_rmsrope_q(const float* __restrict__ qf,
                                                   const float* __restrict__ cosb,
                                                   const float* __restrict__ sinb,
                                                   const float* __restrict__ w,
                                                   unsigned short* __restrict__ qb) {
    int gid = blockIdx.x * 8 + (threadIdx.x >> 5);
    int h = gid % NH_; int bl = gid / NH_;
    int b = bl / L_,  l = bl % L_;
    rms_rope_wave(qf + (size_t)bl * (NH_ * HD_) + h * HD_,
                  cosb + (size_t)l * HD_, sinb + (size_t)l * HD_, w,
                  qb + ((size_t)(b * NH_ + h) * L_ + l) * HD_);
}

__global__ __launch_bounds__(256) void k_rmsrope_k(const float* __restrict__ kf,
                                                   const float* __restrict__ cosb,
                                                   const float* __restrict__ sinb,
                                                   const float* __restrict__ w,
                                                   unsigned short* __restrict__ kb) {
    int gid = blockIdx.x * 8 + (threadIdx.x >> 5);
    int g = gid % NG_; int bl = gid / NG_;
    int b = bl / L_,  l = bl % L_;
    rms_rope_wave(kf + (size_t)bl * (NG_ * HD_) + g * HD_,
                  cosb + (size_t)l * HD_, sinb + (size_t)l * HD_, w,
                  kb + ((size_t)(b * NG_ + g) * L_ + l) * HD_);
}

// ---------------- flash attention (sliding window, GQA) ----------------
#define JT  32
#define LDK 136  // Ks row stride: 128 + 8 pad

__global__ __launch_bounds__(256) void k_attn(const unsigned short* __restrict__ qb,  // [B][NH][L][HD]
                                              const unsigned short* __restrict__ kb,  // [B][NG][L][HD]
                                              const unsigned short* __restrict__ vtb, // [B][NG][HD][L]
                                              unsigned short* __restrict__ ob) {      // [B*L][NH*HD]
    __shared__ __align__(16) unsigned short Ks[2][JT * LDK];   // [j][d]
    __shared__ __align__(16) unsigned short Vs[2][HD_ * LDT];  // [d][j]
    __shared__ __align__(16) unsigned short Pw[8][16 * LDT];   // per-wave P bounce

    const int t = threadIdx.x, wid = t >> 5, lane = t & 31;
    const int hh = lane >> 4, nn = lane & 15;
    const int w0 = __builtin_amdgcn_readfirstlane(t) >> 5;

    int bid = blockIdx.x;
    const int nqt = L_ / 128;
    const int qt = bid % nqt; bid /= nqt;
    const int h  = bid % NH_; bid /= NH_;
    const int b  = bid;
    const int g  = h / GS_;
    const int q0 = qt * 128;
    const int qrow = q0 + wid * 16;

    const unsigned short* qptr = qb + ((size_t)(b * NH_ + h) * L_ + qrow) * HD_;
    const unsigned short* kptr = kb + (size_t)(b * NG_ + g) * L_ * HD_;
    const unsigned short* vptr = vtb + (size_t)(b * NG_ + g) * HD_ * L_;

    const unsigned ks_lds = (unsigned)(uintptr_t)(void*)&Ks[0][0];
    const unsigned vs_lds = (unsigned)(uintptr_t)(void*)&Vs[0][0];

    v16bf qa[4];
#pragma unroll
    for (int dk = 0; dk < 4; ++dk) qa[dk] = load_afrag(qptr + dk * 32, HD_);

    v8f o[8];
    float mrow[8], lrow[8];
#pragma unroll
    for (int nb = 0; nb < 8; ++nb)
#pragma unroll
        for (int e = 0; e < 8; ++e) o[nb][e] = 0.f;
#pragma unroll
    for (int r = 0; r < 8; ++r) { mrow[r] = -1e30f; lrow[r] = 0.f; }

    int js = q0 - WIN_ + 1; if (js < 0) js = 0; js &= ~(JT - 1);
    const int ntiles = (q0 + 128 - js) / JT;

    if (w0 == 0) {  // prologue DMA: tile 0 -> buffer 0
        tdm_load_2d(ks_lds, kptr + (size_t)js * HD_, HD_, JT, HD_, TDM_FLAGS_W128);
        tdm_load_2d(vs_lds, vptr + js, JT, HD_, L_, TDM_FLAGS_W32);
    }

    int cur = 0;
    for (int ti = 0; ti < ntiles; ++ti) {
        const int jt = js + ti * JT;
        if (w0 == 0) {
            if (ti + 1 < ntiles) {
                const int jn = jt + JT;
                tdm_load_2d(ks_lds + (cur ^ 1) * (JT * LDK * 2), kptr + (size_t)jn * HD_,
                            HD_, JT, HD_, TDM_FLAGS_W128);
                tdm_load_2d(vs_lds + (cur ^ 1) * (HD_ * LDT * 2), vptr + jn,
                            JT, HD_, L_, TDM_FLAGS_W32);
                __builtin_amdgcn_s_wait_tensorcnt(2);
            } else {
                __builtin_amdgcn_s_wait_tensorcnt(0);
            }
        }
        __syncthreads();

        const unsigned short* ks = Ks[cur];
        const unsigned short* vs = Vs[cur];

        // S = q . k^T  (two 16x16 j sub-tiles, contracted over d=128)
        v8f sf0, sf1;
#pragma unroll
        for (int e = 0; e < 8; ++e) { sf0[e] = 0.f; sf1[e] = 0.f; }
#pragma unroll
        for (int dk = 0; dk < 4; ++dk) {
            sf0 = wmma_bf16(qa[dk], load_bfrag(ks + 0 * 16 * LDK + dk * 32, LDK), sf0);
            sf1 = wmma_bf16(qa[dk], load_bfrag(ks + 1 * 16 * LDK + dk * 32, LDK), sf1);
        }

        // online softmax per row r (row = qrow + r + 8*hh)
        float sc[8];
#pragma unroll
        for (int r = 0; r < 8; ++r) {
            int ig = qrow + r + 8 * hh;
            int j0 = jt + nn, j1 = j0 + 16;
            bool a0 = (j0 <= ig) && (j0 > ig - WIN_);
            bool a1 = (j1 <= ig) && (j1 > ig - WIN_);
            float v0 = a0 ? sf0[r] * SCALE_ : -1e30f;
            float v1 = a1 ? sf1[r] * SCALE_ : -1e30f;
            float mx = fmaxf(v0, v1);
#pragma unroll
            for (int off = 8; off >= 1; off >>= 1) mx = fmaxf(mx, __shfl_xor(mx, off));
            float mnew = fmaxf(mrow[r], mx);
            float scr = __expf(mrow[r] - mnew);
            float e0 = a0 ? __expf(v0 - mnew) : 0.f;
            float e1 = a1 ? __expf(v1 - mnew) : 0.f;
            float sum = e0 + e1;
#pragma unroll
            for (int off = 8; off >= 1; off >>= 1) sum += __shfl_xor(sum, off);
            lrow[r] = lrow[r] * scr + sum;
            mrow[r] = mnew;
            sc[r] = scr;
            Pw[wid][(r + 8 * hh) * LDT + nn]      = f2bf(e0);
            Pw[wid][(r + 8 * hh) * LDT + nn + 16] = f2bf(e1);
        }

        // O = O*diag(sc) + P . V  (8 d-blocks of 16)
        v16bf ap = load_afrag(&Pw[wid][0], LDT);
#pragma unroll
        for (int nb = 0; nb < 8; ++nb) {
#pragma unroll
            for (int r = 0; r < 8; ++r) o[nb][r] *= sc[r];
            o[nb] = wmma_bf16(ap, load_bfrag(vs + nb * 16 * LDT, LDT), o[nb]);
        }

        __syncthreads();   // reads of buffer `cur` complete
        cur ^= 1;
    }

    // epilogue: O/l -> ob [B*L][NH*HD] bf16 (32-bit offsets)
    const unsigned obase = (unsigned)((b * L_ + qrow + 8 * hh) * (NH_ * HD_) + h * HD_ + nn);
#pragma unroll
    for (int r = 0; r < 8; ++r) {
        float invl = 1.f / lrow[r];
#pragma unroll
        for (int nb = 0; nb < 8; ++nb)
            ob[obase + (unsigned)(r * (NH_ * HD_) + nb * 16)] = f2bf(o[nb][r] * invl);
    }
}

// ---------------- host launcher ----------------
extern "C" void kernel_launch(void* const* d_in, const int* in_sizes, int n_in,
                              void* d_out, int out_size, void* d_ws, size_t ws_size,
                              hipStream_t stream) {
    const float* x    = (const float*)d_in[0];
    const float* cosb = (const float*)d_in[1];
    const float* sinb = (const float*)d_in[2];
    const float* Wq   = (const float*)d_in[3];
    const float* Wk   = (const float*)d_in[4];
    const float* Wv   = (const float*)d_in[5];
    const float* Wo   = (const float*)d_in[6];
    const float* qnw  = (const float*)d_in[7];
    const float* knw  = (const float*)d_in[8];
    float* out = (float*)d_out;

    char* ws = (char*)d_ws;
    unsigned short* xb  = (unsigned short*)(ws);              //  33,554,432
    unsigned short* Wqt = (unsigned short*)(ws +  33554432);  //   8,388,608
    unsigned short* Wkt = (unsigned short*)(ws +  41943040);  //   2,097,152
    unsigned short* Wvt = (unsigned short*)(ws +  44040192);  //   2,097,152
    unsigned short* Wot = (unsigned short*)(ws +  46137344);  //   8,388,608
    float*          qf  = (float*)         (ws +  54525952);  //  67,108,864
    float*          kf  = (float*)         (ws + 121634816);  //  16,777,216
    float*          vf  = (float*)         (ws + 138412032);  //  16,777,216
    unsigned short* qbb = (unsigned short*)(ws + 155189248);  //  33,554,432
    unsigned short* kbb = (unsigned short*)(ws + 188743680);  //   8,388,608
    unsigned short* vtb = (unsigned short*)(ws + 197132288);  //   8,388,608
    unsigned short* ob  = (unsigned short*)(ws + 205520896);  //  33,554,432

    // 1) bf16 conversions; weights transposed with tiled (coalesced) transpose
    k_f2bf<<<(BL_ * IDIM_) / 256, 256, 0, stream>>>(x, xb, BL_ * IDIM_);
    k_transpose_f2bf<<<dim3(32, 32), 256, 0, stream>>>(Wq, Wqt, IDIM_, 2048);
    k_transpose_f2bf<<<dim3(8,  32), 256, 0, stream>>>(Wk, Wkt, IDIM_, 512);
    k_transpose_f2bf<<<dim3(8,  32), 256, 0, stream>>>(Wv, Wvt, IDIM_, 512);
    k_transpose_f2bf<<<dim3(32, 32), 256, 0, stream>>>(Wo, Wot, 2048, IDIM_);

    // 2) projections (TDM double-buffered WMMA GEMM)
    k_gemm_bf16_nt<<<dim3(2048 / BN, BL_ / BM), 256, 0, stream>>>(xb, Wqt, qf, BL_, 2048, IDIM_);
    k_gemm_bf16_nt<<<dim3(512 / BN,  BL_ / BM), 256, 0, stream>>>(xb, Wkt, kf, BL_, 512,  IDIM_);
    k_gemm_bf16_nt<<<dim3(512 / BN,  BL_ / BM), 256, 0, stream>>>(xb, Wvt, vf, BL_, 512,  IDIM_);

    // 3) RMSNorm + RoPE (q,k); V -> bf16 transposed [B][NG][HD][L]
    k_rmsrope_q<<<(BL_ * NH_) / 8, 256, 0, stream>>>(qf, cosb, sinb, qnw, qbb);
    k_rmsrope_k<<<(BL_ * NG_) / 8, 256, 0, stream>>>(kf, cosb, sinb, knw, kbb);
    k_transpose_f2bf<<<dim3(8, 64), 256, 0, stream>>>(vf, vtb, L_, NG_ * HD_);
    k_transpose_f2bf<<<dim3(8, 64), 256, 0, stream>>>(vf + (size_t)L_ * NG_ * HD_,
                                                      vtb + (size_t)NG_ * HD_ * L_,
                                                      L_, NG_ * HD_);

    // 4) sliding-window flash attention (TDM double-buffered)
    k_attn<<<B_ * NH_ * (L_ / 128), 256, 0, stream>>>(qbb, kbb, vtb, ob);

    // 5) output projection
    k_gemm_bf16_nt<<<dim3(IDIM_ / BN, BL_ / BM), 256, 0, stream>>>(ob, Wot, out, BL_, IDIM_, 2048);
}
